// DAMIC_88235808129606
// MI455X (gfx1250) — compile-verified, matrix operand order checked by reading, not run on hardware
//
#include <hip/hip_runtime.h>
#include <math.h>

// ---------------------------------------------------------------------------
// Types
// ---------------------------------------------------------------------------
typedef __bf16 bf16;
typedef __attribute__((ext_vector_type(16))) __bf16    v16bf;
typedef __attribute__((ext_vector_type(8)))  float     v8f;
typedef __attribute__((ext_vector_type(4)))  unsigned  v4u;
typedef __attribute__((ext_vector_type(8)))  int       v8i;
typedef __attribute__((ext_vector_type(4)))  int       v4i;
typedef __attribute__((ext_vector_type(4)))  unsigned  uvec4;

union FragU { uvec4 u[2]; v16bf v; };

// ---------------------------------------------------------------------------
// Problem constants
// ---------------------------------------------------------------------------
#define BATCH 64
#define TLEN  256
#define HDIM  1024
#define GDIM  (4*HDIM)
#define OUTD  12
#define CHUNK 128                 // K-chunk per staging step
#define NBLK  32                  // 32 WGs: each owns 32 hidden cols x 4 gates
#define LSTM_THREADS 256          // 8 waves
#define STAGE_A_BYTES (BATCH*CHUNK*2)   // 16 KB: h tile [64 x 128] bf16
#define STAGE_B_BYTES (4*32*CHUNK*2)    // 32 KB: weights [4 gates x 32 rows x 128] bf16
#define STAGE_BYTES   (STAGE_A_BYTES + STAGE_B_BYTES)
#define LSTM_SMEM     (2*STAGE_BYTES)   // double-buffered: 96 KB of 320 KB/WGP

__constant__ float c_thresh[OUTD] = {
    0.48970118f, 0.34025221f, 0.17722994f, 0.32379692f, 0.1723266f, 0.33252252f,
    0.26682911f, 0.26431107f, 0.2005045f,  0.22233647f, 0.50928269f, 0.35311607f};

#if __has_builtin(__builtin_amdgcn_tensor_load_to_lds)
#define HAVE_TDM 1
#else
#define HAVE_TDM 0
#endif

// ---------------------------------------------------------------------------
// TDM descriptor build + issue (D# per CDNA5 ISA)
// 2-byte elements; 2D tile if tile2==0, else 3D tile (z = gate blocks).
// ---------------------------------------------------------------------------
#if HAVE_TDM
__device__ __forceinline__ void tdm_issue(unsigned lds_off, const void* gptr,
                                          unsigned tile0, unsigned tile1, unsigned tile2,
                                          unsigned long long stride0,
                                          unsigned long long stride1)
{
  unsigned long long ga = (unsigned long long)gptr;
  const unsigned td0 = 1u << 22, td1 = 1u << 22;   // generous dims: OOB never clips
  v4u g0;
  g0[0] = 1u;                                       // count=1 (valid user descriptor)
  g0[1] = lds_off;                                  // lds_addr (bytes)
  g0[2] = (unsigned)ga;                             // global_addr[31:0]
  g0[3] = (unsigned)((ga >> 32) & 0x1FFFFFFull)     // global_addr[56:32]
        | (2u << 30);                               // type = 2 ("image")
  v8i g1;
  g1[0] = (int)(1u << 16);                          // data_size = 1 -> 2 bytes/elem
  g1[1] = (int)((td0 & 0xFFFFu) << 16);             // tensor_dim0[15:0]
  g1[2] = (int)(((td0 >> 16) & 0xFFFFu) | ((td1 & 0xFFFFu) << 16));
  g1[3] = (int)(((td1 >> 16) & 0xFFFFu) | ((tile0 & 0xFFFFu) << 16));
  g1[4] = (int)((tile1 & 0xFFFFu) | ((tile2 & 0xFFFFu) << 16));
  g1[5] = (int)(unsigned)(stride0 & 0xFFFFFFFFull); // tensor_dim0_stride[31:0]
  g1[6] = (int)((unsigned)((stride0 >> 32) & 0xFFFFull)
        | ((unsigned)(stride1 & 0xFFFFull) << 16)); // dim1_stride[15:0]
  g1[7] = (int)(unsigned)((stride1 >> 16) & 0xFFFFFFFFull);
  v4i g2; g2[0] = 64; g2[1] = 0; g2[2] = 0; g2[3] = 0;   // tensor_dim2 bound for z
  v4i g3; g3[0] = 0;  g3[1] = 0; g3[2] = 0; g3[3] = 0;
#if defined(__clang_major__) && (__clang_major__ >= 23)
  v8i g4 = {0,0,0,0,0,0,0,0};
  __builtin_amdgcn_tensor_load_to_lds(g0, g1, g2, g3, g4, 0);
#else
  __builtin_amdgcn_tensor_load_to_lds(g0, g1, g2, g3, 0);
#endif
}
#endif

// ---------------------------------------------------------------------------
// Device-wide barrier for the persistent per-step scan
// ---------------------------------------------------------------------------
__device__ __forceinline__ void grid_barrier(unsigned* arrive, unsigned* epoch,
                                             unsigned nblk, unsigned target)
{
  __syncthreads();
  if (threadIdx.x == 0) {
    unsigned prev = __hip_atomic_fetch_add(arrive, 1u, __ATOMIC_ACQ_REL,
                                           __HIP_MEMORY_SCOPE_AGENT);
    if (prev == nblk - 1u) {
      __hip_atomic_store(arrive, 0u, __ATOMIC_RELAXED, __HIP_MEMORY_SCOPE_AGENT);
      __hip_atomic_fetch_add(epoch, 1u, __ATOMIC_ACQ_REL, __HIP_MEMORY_SCOPE_AGENT);
    } else {
      while (__hip_atomic_load(epoch, __ATOMIC_ACQUIRE, __HIP_MEMORY_SCOPE_AGENT) < target)
        __builtin_amdgcn_s_sleep(2);
    }
  }
  __syncthreads();
}

// ---------------------------------------------------------------------------
// Small prep kernels
// ---------------------------------------------------------------------------
__global__ void zero_kernel(unsigned* __restrict__ p, int nwords) {
  int i = blockIdx.x * blockDim.x + threadIdx.x;
  if (i < nwords) p[i] = 0u;
}

__global__ void f32_to_bf16_kernel(const float* __restrict__ src,
                                   bf16* __restrict__ dst, int n) {
  int i = blockIdx.x * blockDim.x + threadIdx.x;
  if (i < n) dst[i] = (bf16)src[i];
}

// r_logits [B,T,12] -> 12-bit mask stored t-major: mask[t*B + b]
__global__ void pack_mask_kernel(const float* __restrict__ x,
                                 unsigned* __restrict__ mask) {
  int bt = blockIdx.x * blockDim.x + threadIdx.x;
  if (bt >= BATCH * TLEN) return;
  int b = bt / TLEN, t = bt % TLEN;
  const float* p = x + (size_t)bt * OUTD;
  unsigned m = 0u;
  #pragma unroll
  for (int j = 0; j < OUTD; ++j) m |= (p[j] > c_thresh[j]) ? (1u << j) : 0u;
  mask[t * BATCH + b] = m;
}

// ---------------------------------------------------------------------------
// Persistent LSTM layer scan.
// Grid: 32 WGs x 256 thr. WG b owns hidden cols [32b, 32b+32) across 4 gates.
// Wave w: mtile = w&3 (16 batch rows), htile = w>>2 (16 hidden cols),
// 4 accumulators (i,f,g,o). K loop staged through LDS in CHUNK=128 pieces,
// double-buffered via TDM; chunks 0..7 = recurrent (Wrec x h_prev),
// chunks 8..15 (layer1 only) = input (Winp x h0[t]).
// ---------------------------------------------------------------------------
template<bool IS_L0>
__global__ __launch_bounds__(LSTM_THREADS)
void lstm_kernel(const bf16* __restrict__ Wrec, const bf16* __restrict__ Winp,
                 const float* __restrict__ Wih0, const float* __restrict__ bias,
                 const unsigned* __restrict__ mask,
                 const bf16* __restrict__ h_in_seq, bf16* __restrict__ h_out_seq,
                 const bf16* __restrict__ zero_page, float* __restrict__ c_state,
                 unsigned* arrive, unsigned* epoch, int reverse)
{
  extern __shared__ char smem[];
  const int tid   = threadIdx.x;
  const int wid   = tid >> 5;
  const int lane  = tid & 31;
  const int mtile = wid & 3;
  const int htile = wid >> 2;
  const int lrow  = lane & 15;
  const int lhk   = (lane >> 4) * 8;          // per-half-wave K sub-offset
  const int h_local = htile * 16 + lrow;      // 0..31 within WG slice
  const int h_glob  = blockIdx.x * 32 + h_local;
  const int m0      = mtile * 16 + (lane >> 4) * 8;  // batch row of acc elem 0

  const unsigned lds_base = (unsigned)(unsigned long long)smem;
  (void)lds_base;

  // Hoisted per-lane constants (h_glob fixed for whole kernel)
  float bias4[4];
  float w0r[4][OUTD];
  #pragma unroll
  for (int g = 0; g < 4; ++g) {
    bias4[g] = bias[g * HDIM + h_glob];
    if (IS_L0) {
      #pragma unroll
      for (int j = 0; j < OUTD; ++j)
        w0r[g][j] = Wih0[(size_t)(g * HDIM + h_glob) * OUTD + j];
    }
  }

  const int NC = IS_L0 ? (HDIM / CHUNK) : (2 * HDIM / CHUNK);

  for (int s = 0; s < TLEN; ++s) {
    const int t      = reverse ? (TLEN - 1 - s) : s;
    const int prev_t = reverse ? (t + 1) : (t - 1);
    const bf16* h_rec = (s == 0) ? zero_page
                                 : (h_out_seq + (size_t)prev_t * BATCH * HDIM);

    v8f acc[4];
    #pragma unroll
    for (int g = 0; g < 4; ++g)
      acc[g] = (v8f){0.f,0.f,0.f,0.f,0.f,0.f,0.f,0.f};

#if HAVE_TDM
    if (wid == 0) {
      // chunk 0: weights (3D tile: 4 gate blocks of 32 rows) + h tile (2D)
      tdm_issue(lds_base + STAGE_A_BYTES,
                Wrec + (size_t)blockIdx.x * 32 * HDIM,
                CHUNK, 32, 4,
                (unsigned long long)HDIM,
                (unsigned long long)HDIM * (unsigned long long)HDIM);
      tdm_issue(lds_base, h_rec, CHUNK, BATCH, 0,
                (unsigned long long)HDIM, 0ull);
    }
#endif

    for (int c = 0; c < NC; ++c) {
#if HAVE_TDM
      if (wid == 0) {
        if (c + 1 < NC) {
          const int cn = c + 1;
          const size_t k0 = (size_t)(cn & 7) * CHUNK;
          const bf16* wsrc = (cn < 8) ? Wrec : Winp;
          const bf16* asrc = (cn < 8)
              ? (h_rec + k0)
              : (h_in_seq + (size_t)t * BATCH * HDIM + k0);
          const unsigned sb = lds_base + (unsigned)(cn & 1) * STAGE_BYTES;
          tdm_issue(sb + STAGE_A_BYTES,
                    wsrc + (size_t)blockIdx.x * 32 * HDIM + k0,
                    CHUNK, 32, 4,
                    (unsigned long long)HDIM,
                    (unsigned long long)HDIM * (unsigned long long)HDIM);
          tdm_issue(sb, asrc, CHUNK, BATCH, 0,
                    (unsigned long long)HDIM, 0ull);
          __builtin_amdgcn_s_wait_tensorcnt(2);  // chunk c complete (in-order)
        } else {
          __builtin_amdgcn_s_wait_tensorcnt(0);
        }
      }
      __syncthreads();
#else
      // Fallback: cooperative block staging (no TDM builtin on this toolchain)
      {
        const size_t k0 = (size_t)(c & 7) * CHUNK;
        const bf16* wsrc = (c < 8) ? Wrec : Winp;
        const bf16* asrc = (c < 8)
            ? (h_rec + k0)
            : (h_in_seq + (size_t)t * BATCH * HDIM + k0);
        char* ab = smem + (c & 1) * STAGE_BYTES;
        char* bb = ab + STAGE_A_BYTES;
        // A: 64 rows x 128 elems = 1024 x 16B
        for (int i = tid; i < BATCH * CHUNK / 8; i += LSTM_THREADS) {
          int row = i / (CHUNK / 8), q = i % (CHUNK / 8);
          *(uvec4*)(ab + (size_t)i * 16) =
              *(const uvec4*)(asrc + (size_t)row * HDIM + q * 8);
        }
        // B: 4*32 rows x 128 elems = 2048 x 16B
        for (int i = tid; i < 4 * 32 * CHUNK / 8; i += LSTM_THREADS) {
          int r = i / (CHUNK / 8), q = i % (CHUNK / 8);
          int gate = r / 32, row = r % 32;
          *(uvec4*)(bb + (size_t)i * 16) =
              *(const uvec4*)(wsrc + ((size_t)gate * HDIM + blockIdx.x * 32 + row) * HDIM
                              + k0 + q * 8);
        }
      }
      __syncthreads();
#endif

      const char* ab = smem + (c & 1) * STAGE_BYTES;
      const char* bb = ab + STAGE_A_BYTES;
      // Load A + all 4 gate B fragments for a K-step first, then issue the
      // 4 WMMAs back-to-back: lets the scheduler hoist the next K-step's
      // ds_loads above the current WMMAs instead of draining DScnt per op.
      #pragma unroll
      for (int kk = 0; kk < CHUNK; kk += 32) {
        FragU afr;
        const unsigned aoff = (unsigned)(((mtile * 16 + lrow) * CHUNK) + kk + lhk) * 2u;
        afr.u[0] = *(const uvec4*)(ab + aoff);
        afr.u[1] = *(const uvec4*)(ab + aoff + 32);
        FragU bfr[4];
        #pragma unroll
        for (int g = 0; g < 4; ++g) {
          const unsigned boff = (unsigned)((((g * 32 + h_local) * CHUNK) + kk + lhk)) * 2u;
          bfr[g].u[0] = *(const uvec4*)(bb + boff);
          bfr[g].u[1] = *(const uvec4*)(bb + boff + 32);
        }
        #pragma unroll
        for (int g = 0; g < 4; ++g) {
          acc[g] = __builtin_amdgcn_wmma_f32_16x16x32_bf16(
                     false, afr.v, false, bfr[g].v, (short)0, acc[g], false, false);
        }
      }
      __syncthreads();
    }

    // -------- gate nonlinearities + cell/hidden update (wave-local) --------
    unsigned mk[8] = {0,0,0,0,0,0,0,0};
    if (IS_L0) {
      #pragma unroll
      for (int v = 0; v < 8; ++v) mk[v] = mask[t * BATCH + m0 + v];
    }
    #pragma unroll
    for (int v = 0; v < 8; ++v) {
      const int batch = m0 + v;
      float gate[4];
      #pragma unroll
      for (int g = 0; g < 4; ++g) {
        float x = acc[g][v] + bias4[g];
        if (IS_L0) {
          #pragma unroll
          for (int j = 0; j < OUTD; ++j)
            x += ((mk[v] >> j) & 1u) ? w0r[g][j] : 0.f;
        }
        gate[g] = x;
      }
      const size_t ci = (size_t)batch * HDIM + h_glob;
      const float cprev = c_state[ci];
      const float gi = 1.f / (1.f + __expf(-gate[0]));
      const float gf = 1.f / (1.f + __expf(-gate[1]));
      const float gg = tanhf(gate[2]);
      const float go = 1.f / (1.f + __expf(-gate[3]));
      const float cn = gf * cprev + gi * gg;
      const float hn = go * tanhf(cn);
      c_state[ci] = cn;
      h_out_seq[(size_t)t * BATCH * HDIM + ci] = (bf16)hn;
    }

    grid_barrier(arrive, epoch, gridDim.x, (unsigned)(s + 1));
  }
}

// ---------------------------------------------------------------------------
// Output projection: out[b,t,o] = sigmoid(bout[o] + rev.Wout[:, :H] + fwd.Wout[:, H:])
// one wave per (b,t); K = 2x1024 bf16 reads, f32 accumulate, lane reduction.
// ---------------------------------------------------------------------------
__global__ __launch_bounds__(256)
void proj_kernel(const bf16* __restrict__ rev1, const bf16* __restrict__ fwd1,
                 const float* __restrict__ Wout, const float* __restrict__ bout,
                 float* __restrict__ out)
{
  const int wid  = threadIdx.x >> 5;
  const int lane = threadIdx.x & 31;
  const int idx  = blockIdx.x * 8 + wid;       // 0 .. B*T-1
  const int b = idx / TLEN, t = idx % TLEN;
  const bf16* r = rev1 + ((size_t)t * BATCH + b) * HDIM;
  const bf16* f = fwd1 + ((size_t)t * BATCH + b) * HDIM;
  float acc[OUTD];
  #pragma unroll
  for (int o = 0; o < OUTD; ++o) acc[o] = 0.f;
  for (int h = lane; h < HDIM; h += 32) {
    const float rv = (float)r[h];
    const float fv = (float)f[h];
    #pragma unroll
    for (int o = 0; o < OUTD; ++o)
      acc[o] += rv * Wout[(size_t)o * 2 * HDIM + h]
              + fv * Wout[(size_t)o * 2 * HDIM + HDIM + h];
  }
  #pragma unroll
  for (int o = 0; o < OUTD; ++o) {
    #pragma unroll
    for (int d = 16; d > 0; d >>= 1) acc[o] += __shfl_xor(acc[o], d, 32);
  }
  if (lane == 0) {
    #pragma unroll
    for (int o = 0; o < OUTD; ++o)
      out[((size_t)b * TLEN + t) * OUTD + o] =
          1.f / (1.f + __expf(-(acc[o] + bout[o])));
  }
}

// ---------------------------------------------------------------------------
// Host launch
// ---------------------------------------------------------------------------
extern "C" void kernel_launch(void* const* d_in, const int* in_sizes, int n_in,
                              void* d_out, int out_size, void* d_ws, size_t ws_size,
                              hipStream_t stream) {
  (void)in_sizes; (void)n_in; (void)out_size; (void)ws_size;
  char* w = (char*)d_ws;

  size_t off = 0;
  auto take = [&](size_t bytes) { size_t r = off; off = (off + bytes + 255) & ~(size_t)255; return r; };

  const size_t sync_off = take(1024);
  const size_t zp_off   = take((size_t)BATCH * HDIM * 2);     // zero h page (bf16)
  size_t c_off[4];
  for (int i = 0; i < 4; ++i) c_off[i] = take((size_t)BATCH * HDIM * 4);
  const size_t zero_end = off;                                // zero-init region
  size_t w_off[6];
  for (int i = 0; i < 6; ++i) w_off[i] = take((size_t)GDIM * HDIM * 2);  // bf16 weights
  const size_t mask_off = take((size_t)TLEN * BATCH * 4);
  const size_t h0_off   = take((size_t)TLEN * BATCH * HDIM * 2);  // layer-0 output (reused per dir)
  const size_t rev1_off = take((size_t)TLEN * BATCH * HDIM * 2);
  const size_t fwd1_off = take((size_t)TLEN * BATCH * HDIM * 2);

  // 1) zero sync counters, zero page, cell states
  {
    int nwords = (int)(zero_end / 4);
    zero_kernel<<<(nwords + 255) / 256, 256, 0, stream>>>((unsigned*)w, nwords);
  }
  // 2) fp32 -> bf16 weight conversion (weights then live in the 192MB L2)
  {
    const int widx[6] = {2, 4, 5, 8, 10, 11};  // rWhh0,rWih1,rWhh1,fWhh0,fWih1,fWhh1
    const int n = GDIM * HDIM;
    for (int i = 0; i < 6; ++i)
      f32_to_bf16_kernel<<<(n + 255) / 256, 256, 0, stream>>>(
          (const float*)d_in[widx[i]], (bf16*)(w + w_off[i]), n);
  }
  // 3) threshold + pack binary input into 12-bit masks
  pack_mask_kernel<<<(BATCH * TLEN + 255) / 256, 256, 0, stream>>>(
      (const float*)d_in[0], (unsigned*)(w + mask_off));

  unsigned* sync = (unsigned*)(w + sync_off);
  const bf16* zp = (const bf16*)(w + zp_off);

  // 4) four persistent layer scans: rev L0, rev L1, fwd L0, fwd L1
  lstm_kernel<true><<<NBLK, LSTM_THREADS, LSTM_SMEM, stream>>>(
      (const bf16*)(w + w_off[0]), nullptr,
      (const float*)d_in[1], (const float*)d_in[3],
      (const unsigned*)(w + mask_off), nullptr,
      (bf16*)(w + h0_off), zp, (float*)(w + c_off[0]),
      sync + 0, sync + 16, 1);

  lstm_kernel<false><<<NBLK, LSTM_THREADS, LSTM_SMEM, stream>>>(
      (const bf16*)(w + w_off[2]), (const bf16*)(w + w_off[1]),
      nullptr, (const float*)d_in[6],
      nullptr, (const bf16*)(w + h0_off),
      (bf16*)(w + rev1_off), zp, (float*)(w + c_off[1]),
      sync + 32, sync + 48, 1);

  lstm_kernel<true><<<NBLK, LSTM_THREADS, LSTM_SMEM, stream>>>(
      (const bf16*)(w + w_off[3]), nullptr,
      (const float*)d_in[7], (const float*)d_in[9],
      (const unsigned*)(w + mask_off), nullptr,
      (bf16*)(w + h0_off), zp, (float*)(w + c_off[2]),
      sync + 64, sync + 80, 0);

  lstm_kernel<false><<<NBLK, LSTM_THREADS, LSTM_SMEM, stream>>>(
      (const bf16*)(w + w_off[5]), (const bf16*)(w + w_off[4]),
      nullptr, (const float*)d_in[12],
      nullptr, (const bf16*)(w + h0_off),
      (bf16*)(w + fwd1_off), zp, (float*)(w + c_off[3]),
      sync + 96, sync + 112, 0);

  // 5) final projection + sigmoid
  proj_kernel<<<(BATCH * TLEN) / 8, 256, 0, stream>>>(
      (const bf16*)(w + rev1_off), (const bf16*)(w + fwd1_off),
      (const float*)d_in[13], (const float*)d_in[14], (float*)d_out);
}
